// ViTSelfAttention_16243566313558
// MI455X (gfx1250) — matrix operand order repaired and checked by reading, not compiled, not run on hardware
//
#include <hip/hip_runtime.h>
#include <hip/hip_bf16.h>

// ---------------------------------------------------------------------------
// ViT self-attention forward, f16 WMMA pipeline for gfx1250 (MI455X).
// B=16, N=1024, D=768, H=12, HD=64.  SCALE = 1/8.
// NOTE: vit_attn is defined FIRST so the disasm snippet shows its codegen
// (TDM issue / s_wait_tensorcnt / ds-fed WMMA loop).
// ---------------------------------------------------------------------------

typedef __attribute__((ext_vector_type(16))) _Float16     v16h;
typedef __attribute__((ext_vector_type(8)))  _Float16     v8h;
typedef __attribute__((ext_vector_type(8)))  float        v8f;
typedef __attribute__((ext_vector_type(4)))  unsigned int u32x4;
typedef __attribute__((ext_vector_type(8)))  int          i32x8;
typedef __attribute__((ext_vector_type(4)))  int          i32x4;

#define WMMA_F32_F16(A, B, C) \
    __builtin_amdgcn_wmma_f32_16x16x32_f16(false, (A), false, (B), (short)0, (C), false, false)

#if __has_builtin(__builtin_amdgcn_tensor_load_to_lds) && \
    __has_builtin(__builtin_amdgcn_s_wait_tensorcnt)
#define HAVE_TDM 1
#else
#define HAVE_TDM 0
#endif

// ----- A-matrix tile (16x32 f16) from row-major f16 src -------------------
// ISA 7.12.2: lanes 0-15 hold row M=lane, K=0..7 / K=16..23;
//             lanes 16-31 hold row M=lane-16, K=8..15 / K=24..31
__device__ inline v16h load_a16(const _Float16* base, int ld,
                                int row0, int col0, int lane) {
    const int r  = row0 + (lane & 15);
    const int kb = col0 + ((lane >> 4) << 3);
    const v8h lo = *(const v8h*)(base + (size_t)r * ld + kb);
    const v8h hi = *(const v8h*)(base + (size_t)r * ld + kb + 16);
    return __builtin_shufflevector(lo, hi, 0,1,2,3,4,5,6,7,8,9,10,11,12,13,14,15);
}

// ----- B-matrix tile (32x16 f16) from column-contiguous (B^T row-major) ---
// lanes 0-15: column N=lane, K=0..7 / K=16..23; lanes 16-31: N=lane-16, K=8..15 / K=24..31
__device__ inline v16h load_bT16(const _Float16* baseT, int ld,
                                 int k0, int col0, int lane) {
    const int n  = col0 + (lane & 15);
    const int kb = k0 + ((lane >> 4) << 3);
    const v8h lo = *(const v8h*)(baseT + (size_t)n * ld + kb);
    const v8h hi = *(const v8h*)(baseT + (size_t)n * ld + kb + 16);
    return __builtin_shufflevector(lo, hi, 0,1,2,3,4,5,6,7,8,9,10,11,12,13,14,15);
}

// ----- A-matrix tile from f32 LDS with convert-on-load --------------------
__device__ inline v16h load_aF32(const float* s, int ld, int col0, int lane) {
    const int r  = lane & 15;
    const int kb = col0 + ((lane >> 4) << 3);
    const float* p = s + r * ld + kb;
    v16h a;
#pragma unroll
    for (int i = 0; i < 8; ++i) a[i]     = (_Float16)p[i];
#pragma unroll
    for (int i = 0; i < 8; ++i) a[8 + i] = (_Float16)p[16 + i];
    return a;
}

// ----- Tensor Data Mover: 64x64 f16 tile global -> LDS, rows padded to 72 -
// D# per CDNA5 ISA §8: group0 {count=1, lds_addr, global_addr, type=2},
// group1 {data_size=2B, pad_enable, pad_interval=32DW, pad_amount=4DW, dims}.
__device__ inline void tdm_load_64x64_f16(unsigned lds_byte_addr, const _Float16* gsrc,
                                          unsigned stride_elems) {
#if HAVE_TDM
    unsigned long long ga = (unsigned long long)(uintptr_t)gsrc;
    u32x4 g0 = {};
    g0[0] = 1u;                                           // count=1, user mode
    g0[1] = lds_byte_addr;                                // lds_addr
    g0[2] = (unsigned)(ga & 0xffffffffull);               // global_addr[31:0]
    g0[3] = (unsigned)((ga >> 32) & 0x01ffffffull) | (2u << 30);  // addr hi | type=2
    i32x8 g1 = {};
    g1[0] = (int)((1u << 16) | (1u << 20) | (4u << 22) | (3u << 25));
    g1[1] = (int)(64u << 16);                             // tensor_dim0 = 64
    g1[2] = (int)(64u << 16);                             // tensor_dim1 = 64 (lo16)
    g1[3] = (int)(64u << 16);                             // tile_dim0 = 64
    g1[4] = (int)64;                                      // tile_dim1 = 64 (tile_dim2=0)
    g1[5] = (int)stride_elems;                            // tensor_dim0_stride
    g1[6] = 0;
    g1[7] = 0;
    i32x4 g2 = {};
    i32x4 g3 = {};
#if defined(__clang_major__) && __clang_major__ >= 23
    i32x8 g4 = {};
    __builtin_amdgcn_tensor_load_to_lds(g0, g1, g2, g3, g4, 0);
#else
    __builtin_amdgcn_tensor_load_to_lds(g0, g1, g2, g3, 0);
#endif
#else
    (void)lds_byte_addr; (void)gsrc; (void)stride_elems;
#endif
}

// ---------------------------------------------------------------------------
// Attention kernel (defined first for disasm visibility).
// 2-wave blocks; each block owns a 32-row q-superblock of one (b,h).
// K and V^T are staged as double-buffered 64x64 f16 tiles via TDM; each wave
// keeps a 16x1028 f32 score strip; two-pass softmax; both GEMMs on WMMA.
// ---------------------------------------------------------------------------
#define LDP      1028                       // score strip stride (f32)
#define KCH      64                         // keys per staged chunk
#define LDH      72                         // staged tile LDS row stride (halves)
#define STAGE1_B (KCH * LDH * 2)            //  9,216 B per buffer
#define STAGE_B  (2 * STAGE1_B)             // 18,432 B double buffer
#define STRIP_B  (16 * LDP * 4)             // 65,792 B per wave
#define INV_OFF  (STAGE_B + 2 * STRIP_B)    // 150,016
#define SMEM_B   (INV_OFF + 128)            // 150,144 B total

__global__ void vit_attn(const _Float16* __restrict__ Qh,
                         const _Float16* __restrict__ Kh,
                         const _Float16* __restrict__ Vt,
                         float* __restrict__ out) {
    extern __shared__ __align__(16) char smem_raw[];
    _Float16* stage0 = (_Float16*)(smem_raw);
    _Float16* stage1 = (_Float16*)(smem_raw + STAGE1_B);
    const int wid  = threadIdx.x >> 5;
    const int lane = threadIdx.x & 31;
    float* S   = (float*)(smem_raw + STAGE_B + wid * STRIP_B);
    float* inv = (float*)(smem_raw + INV_OFF) + wid * 16;

    const int bh = blockIdx.x >> 5;          // 192 (b,h) pairs
    const int sb = blockIdx.x & 31;          // 32 superblocks of 32 rows
    const int b  = bh / 12;
    const int h  = bh % 12;
    const int q0 = sb * 32 + wid * 16;

    const _Float16* Q = Qh + (size_t)bh * 1024 * 64;
    const _Float16* K = Kh + (size_t)bh * 1024 * 64;
    const _Float16* V = Vt + (size_t)bh * 64 * 1024;

#if HAVE_TDM
    const unsigned ldsStage = (unsigned)__builtin_amdgcn_groupstaticsize();
#endif

    const v16h aq0 = load_a16(Q, 64, q0, 0,  lane);
    const v16h aq1 = load_a16(Q, 64, q0, 32, lane);
    const int colL  = lane & 15;
    const int rbase = (lane >> 4) << 3;

    // ---- phase 1: S = (Q K^T) * scale; K chunks double-buffered via TDM ----
#if HAVE_TDM
    if (wid == 0) tdm_load_64x64_f16(ldsStage, K, 64);    // prefetch chunk 0
#endif
    for (int c = 0; c < 16; ++c) {
        const int c0 = c * KCH;
        _Float16* buf = (c & 1) ? stage1 : stage0;
#if HAVE_TDM
        if (wid == 0) {   // wave 0 drives the DMA (EXEC ignored, TENSORcnt per-wave)
            if (c < 15) {
                tdm_load_64x64_f16(ldsStage + (((c + 1) & 1) ? STAGE1_B : 0),
                                   K + (size_t)(c0 + KCH) * 64, 64);
                __builtin_amdgcn_s_wait_tensorcnt((short)1);   // chunk c done
            } else {
                __builtin_amdgcn_s_wait_tensorcnt((short)0);
            }
        }
#else
        for (int i = threadIdx.x; i < KCH * 8; i += 64) {      // fallback: direct copy
            int row = i >> 3, c8 = (i & 7) << 3;
            *(v8h*)(buf + row * LDH + c8) = *(const v8h*)(K + (size_t)(c0 + row) * 64 + c8);
        }
#endif
        __syncthreads();
#pragma unroll
        for (int nt = 0; nt < KCH / 16; ++nt) {
            v8f c8f = v8f{};
            v16h b0 = load_bT16(buf, LDH, 0,  nt * 16, lane);
            c8f = WMMA_F32_F16(aq0, b0, c8f);
            v16h b1 = load_bT16(buf, LDH, 32, nt * 16, lane);
            c8f = WMMA_F32_F16(aq1, b1, c8f);
#pragma unroll
            for (int r = 0; r < 8; ++r)
                S[(rbase + r) * LDP + c0 + nt * 16 + colL] = c8f[r] * 0.125f;
        }
        __syncthreads();
    }

    // ---- phase 2: per-wave row softmax on own strip ----
    for (int r = 0; r < 16; ++r) {
        float* row = S + r * LDP;
        float m = -3.402823466e38f;
#pragma unroll
        for (int i = 0; i < 32; ++i) m = fmaxf(m, row[lane + i * 32]);
#pragma unroll
        for (int off = 16; off > 0; off >>= 1) m = fmaxf(m, __shfl_xor(m, off, 32));
        float sum = 0.f;
#pragma unroll
        for (int i = 0; i < 32; ++i) {
            float e = __expf(row[lane + i * 32] - m);
            row[lane + i * 32] = e;
            sum += e;
        }
#pragma unroll
        for (int off = 16; off > 0; off >>= 1) sum += __shfl_xor(sum, off, 32);
        if (lane == 0) inv[r] = 1.0f / sum;
    }

    // ---- phase 3: ctx = P V; V^T chunks double-buffered via TDM ----
    v8f acc[4] = {v8f{}, v8f{}, v8f{}, v8f{}};
#if HAVE_TDM
    if (wid == 0) tdm_load_64x64_f16(ldsStage, V, 1024);  // prefetch chunk 0
#endif
    for (int c = 0; c < 16; ++c) {
        const int kk0 = c * KCH;
        _Float16* buf = (c & 1) ? stage1 : stage0;
#if HAVE_TDM
        if (wid == 0) {
            if (c < 15) {
                tdm_load_64x64_f16(ldsStage + (((c + 1) & 1) ? STAGE1_B : 0),
                                   V + kk0 + KCH, 1024);
                __builtin_amdgcn_s_wait_tensorcnt((short)1);
            } else {
                __builtin_amdgcn_s_wait_tensorcnt((short)0);
            }
        }
#else
        for (int i = threadIdx.x; i < 64 * 8; i += 64) {
            int row = i >> 3, c8 = (i & 7) << 3;
            *(v8h*)(buf + row * LDH + c8) = *(const v8h*)(V + (size_t)row * 1024 + kk0 + c8);
        }
#endif
        __syncthreads();
#pragma unroll
        for (int ks = 0; ks < KCH / 32; ++ks) {
            v16h a = load_aF32(S, LDP, kk0 + ks * 32, lane);
#pragma unroll
            for (int t = 0; t < 4; ++t) {
                v16h bm = load_bT16(buf, LDH, ks * 32, t * 16, lane);
                acc[t] = WMMA_F32_F16(a, bm, acc[t]);
            }
        }
        __syncthreads();
    }

#pragma unroll
    for (int t = 0; t < 4; ++t) {
        const int d = t * 16 + colL;
#pragma unroll
        for (int r = 0; r < 8; ++r) {
            const int row = rbase + r;
            const int tok = q0 + row;
            out[((size_t)(b * 1024 + tok)) * 768 + h * 64 + d] = acc[t][r] * inv[row];
        }
    }
}

// ---------------------------------------------------------------------------
// Kernel 1a: x (f32) -> xh (f16)
// ---------------------------------------------------------------------------
__global__ void vit_cvt_x(const float* __restrict__ x, _Float16* __restrict__ xh, int n) {
    int i = blockIdx.x * blockDim.x + threadIdx.x;
    if (i < n) xh[i] = (_Float16)x[i];
}

// ---------------------------------------------------------------------------
// Kernel 1b: Wq/Wk/Wv (f32 [k][n]) -> Wt (f16 [w][n][k])  (transposed)
// ---------------------------------------------------------------------------
__global__ void vit_cvt_w(const float* __restrict__ Wq, const float* __restrict__ Wk,
                          const float* __restrict__ Wv, _Float16* __restrict__ Wt) {
    int idx = blockIdx.x * blockDim.x + threadIdx.x;
    if (idx >= 3 * 768 * 768) return;
    int w   = idx / (768 * 768);
    int rem = idx % (768 * 768);
    int n   = rem / 768;
    int k   = rem % 768;
    const float* W = (w == 0) ? Wq : (w == 1) ? Wk : Wv;
    Wt[idx] = (_Float16)W[k * 768 + n];
}

// ---------------------------------------------------------------------------
// Kernel 2: fused QKV projection GEMM [16384 x 768] x [768 x 2304] + bias.
// One wave -> 16x64 output tile; epilogue scatters to head-major layouts:
//   Qh [bh][tok][64], Kh [bh][tok][64], Vt [bh][64][tok]  (f16)
// ---------------------------------------------------------------------------
__global__ void vit_qkv_gemm(const _Float16* __restrict__ xh,
                             const _Float16* __restrict__ Wt,
                             const float* __restrict__ bq,
                             const float* __restrict__ bk,
                             const float* __restrict__ bv,
                             _Float16* __restrict__ Qh,
                             _Float16* __restrict__ Kh,
                             _Float16* __restrict__ Vt) {
    const int lane = threadIdx.x & 31;
    const int wave = blockIdx.x * (blockDim.x >> 5) + (threadIdx.x >> 5);
    const int m0 = (wave / 36) * 16;
    const int n0 = (wave % 36) * 64;

    v8f acc[4] = {v8f{}, v8f{}, v8f{}, v8f{}};
    for (int kk = 0; kk < 768; kk += 32) {
        if (kk + 32 < 768)
            __builtin_prefetch(xh + (size_t)(m0 + (lane & 15)) * 768 + kk + 32, 0, 3);
        v16h a = load_a16(xh, 768, m0, kk, lane);
#pragma unroll
        for (int t = 0; t < 4; ++t) {
            v16h b = load_bT16(Wt, 768, kk, n0 + t * 16, lane);
            acc[t] = WMMA_F32_F16(a, b, acc[t]);
        }
    }

    const int colL  = lane & 15;
    const int rbase = (lane >> 4) << 3;
#pragma unroll
    for (int t = 0; t < 4; ++t) {
        const int ng = n0 + t * 16 + colL;
        const int w  = ng / 768;
        const int f  = ng % 768;
        const int h  = f >> 6;
        const int d  = f & 63;
        const float bias = ((w == 0) ? bq : (w == 1) ? bk : bv)[f];
#pragma unroll
        for (int r = 0; r < 8; ++r) {
            const int rowm = m0 + rbase + r;
            const int bb   = rowm >> 10;
            const int tok  = rowm & 1023;
            const int bhh  = bb * 12 + h;
            const _Float16 hv = (_Float16)(acc[t][r] + bias);
            if (w == 0)      Qh[((size_t)bhh * 1024 + tok) * 64 + d] = hv;
            else if (w == 1) Kh[((size_t)bhh * 1024 + tok) * 64 + d] = hv;
            else             Vt[((size_t)bhh * 64 + d) * 1024 + tok] = hv;
        }
    }
}

// ---------------------------------------------------------------------------
// Host-side launch
// ---------------------------------------------------------------------------
extern "C" void kernel_launch(void* const* d_in, const int* in_sizes, int n_in,
                              void* d_out, int out_size, void* d_ws, size_t ws_size,
                              hipStream_t stream) {
    const float* x  = (const float*)d_in[0];
    const float* Wq = (const float*)d_in[1];
    const float* bq = (const float*)d_in[2];
    const float* Wk = (const float*)d_in[3];
    const float* bk = (const float*)d_in[4];
    const float* Wv = (const float*)d_in[5];
    const float* bv = (const float*)d_in[6];
    float* out = (float*)d_out;

    // Workspace: xh | Wt | Qh | Kh | Vt  (all 256B aligned by construction)
    char* ws = (char*)d_ws;
    _Float16* xh = (_Float16*)(ws);
    _Float16* Wt = (_Float16*)(ws + 25165824);
    _Float16* Qh = (_Float16*)(ws + 25165824 + 3538944);
    _Float16* Kh = (_Float16*)(ws + 25165824 + 3538944 + 25165824);
    _Float16* Vt = (_Float16*)(ws + 25165824 + 3538944 + 2 * 25165824);

    {
        int n = 16384 * 768;                       // 12,582,912 = 256 * 49152
        vit_cvt_x<<<n / 256, 256, 0, stream>>>(x, xh, n);
        int nw = 3 * 768 * 768;                    //  1,769,472 = 256 * 6912
        vit_cvt_w<<<nw / 256, 256, 0, stream>>>(Wq, Wk, Wv, Wt);
    }

    // QKV GEMM: 1024 row-tiles * 36 col-chunks = 36864 waves, 8 waves/block
    vit_qkv_gemm<<<36864 / 8, 256, 0, stream>>>(xh, Wt, bq, bk, bv, Qh, Kh, Vt);

    // Attention: 192 heads * 32 superblocks = 6144 two-wave blocks
    vit_attn<<<6144, 64, SMEM_B, stream>>>(Qh, Kh, Vt, out);
}
